// MultiHeadAttention_15839839388294
// MI455X (gfx1250) — compile-verified, matrix operand order checked by reading, not run on hardware
//
#include <hip/hip_runtime.h>
#include <hip/hip_bf16.h>

// Problem constants
#define Sq   4096
#define Dm   768
#define Hn   12
#define HDm  64
#define Bn   2
#define BHn  24
#define Mr   8192   // B*S

typedef __attribute__((ext_vector_type(16))) __bf16 v16bf;
typedef __attribute__((ext_vector_type(8)))  __bf16 v8bf;
typedef __attribute__((ext_vector_type(8)))  float  v8f;
typedef __attribute__((ext_vector_type(4)))  float  v4f;
typedef __attribute__((ext_vector_type(4)))  unsigned v4u;
typedef __attribute__((ext_vector_type(8)))  int      v8i;
typedef __attribute__((ext_vector_type(4)))  int      v4i;

#if defined(__HIP_DEVICE_COMPILE__) && __has_builtin(__builtin_amdgcn_tensor_load_to_lds)
#define HAVE_TDM 1
#else
#define HAVE_TDM 0
#endif

__device__ __forceinline__ void wait_tensorcnt0() {
#if defined(__HIP_DEVICE_COMPILE__)
#if __has_builtin(__builtin_amdgcn_s_wait_tensorcnt)
  __builtin_amdgcn_s_wait_tensorcnt(0);
#elif defined(__AMDGCN__)
  asm volatile("s_wait_tensorcnt 0x0" ::: "memory");
#endif
#endif
}

// Flat shared-memory pointer -> LDS byte address (ISA 10.2: LDS aperture maps
// addr[31:0] to the LDS byte offset).
__device__ __forceinline__ unsigned lds_u32(const void* p) {
  return (unsigned)(unsigned long long)p;
}

#if HAVE_TDM
// Issue a 2-D TDM tile load (bf16 elements): tile_d0 contiguous elems per row,
// tile_d1 rows, row stride stride0 (elements). Wave-level op; EXEC ignored.
__device__ __forceinline__ void tdm_load_2d(unsigned lds_off, const void* gptr,
                                            unsigned tile_d0, unsigned tile_d1,
                                            unsigned tensor_d0, unsigned tensor_d1,
                                            unsigned long long stride0) {
  unsigned long long ga = (unsigned long long)gptr;
  v4u g0;
  g0[0] = 1u;                                         // count=1, user mode
  g0[1] = lds_off;                                    // lds_addr[31:0]
  g0[2] = (unsigned)ga;                               // global_addr[31:0]
  g0[3] = (unsigned)((ga >> 32) & 0x01FFFFFFull)      // global_addr[56:32]
          | (2u << 30);                               // type = 2 ("image")
  v8i g1;
  g1[0] = 0x00010000;                                 // data_size=1 -> 2 bytes
  g1[1] = (int)((tensor_d0 & 0xFFFFu) << 16);         // tensor_dim0[15:0]
  g1[2] = (int)(((tensor_d0 >> 16) & 0xFFFFu) |
                ((tensor_d1 & 0xFFFFu) << 16));       // td0[31:16] | td1[15:0]
  g1[3] = (int)(((tensor_d1 >> 16) & 0xFFFFu) |
                ((tile_d0 & 0xFFFFu) << 16));         // td1[31:16] | tile_dim0
  g1[4] = (int)(tile_d1 & 0xFFFFu);                   // tile_dim1 (tile_dim2=0)
  g1[5] = (int)(stride0 & 0xFFFFFFFFull);             // dim0_stride[31:0]
  g1[6] = (int)((stride0 >> 32) & 0xFFFFull);         // dim0_stride[47:32]
  g1[7] = 0;                                          // dim1_stride hi = 0
  v4i gz = {0, 0, 0, 0};
#if __clang_major__ >= 23
  v8i gz8 = {0, 0, 0, 0, 0, 0, 0, 0};
  __builtin_amdgcn_tensor_load_to_lds(g0, g1, gz, gz, gz8, 0);
#else
  __builtin_amdgcn_tensor_load_to_lds(g0, g1, gz, gz, 0);
#endif
}
#endif

__device__ __forceinline__ v8f wmma_bf16(v16bf a, v16bf b, v8f c) {
  return __builtin_amdgcn_wmma_f32_16x16x32_bf16(
      false, a, false, b, (short)0, c, false, false);
}

// WMMA A/B operand (per-lane 16 bf16 along K) from row-major fp32: p[0..7],p[16..23]
__device__ __forceinline__ v16bf ld_op_f32(const float* p) {
  v4f a0 = *(const v4f*)(p);      v4f a1 = *(const v4f*)(p + 4);
  v4f a2 = *(const v4f*)(p + 16); v4f a3 = *(const v4f*)(p + 20);
  v16bf r;
#pragma unroll
  for (int i = 0; i < 4; ++i) {
    r[i]      = (__bf16)a0[i];
    r[i + 4]  = (__bf16)a1[i];
    r[i + 8]  = (__bf16)a2[i];
    r[i + 12] = (__bf16)a3[i];
  }
  return r;
}

// Same but source already bf16 (global or LDS; LDS -> ds_load_b128).
__device__ __forceinline__ v16bf ld_op_bf16(const __bf16* p) {
  v8bf lo = *(const v8bf*)(p);
  v8bf hi = *(const v8bf*)(p + 16);
  v16bf r;
#pragma unroll
  for (int i = 0; i < 8; ++i) { r[i] = lo[i]; r[i + 8] = hi[i]; }
  return r;
}

// ---------------- Kernel 1: fused QKV projection ----------------
// grid: (Mr/64, Dm/64, 3), block 128 (4 waves; wave -> 16x64 output strip)
// z=0 -> Q [B,H,S,64] bf16 ; z=1 -> K [B,H,S,64] bf16 ; z=2 -> V^T [B,H,64,S] bf16
__global__ void __launch_bounds__(128) qkv_proj_kernel(
    const float* __restrict__ q_in, const float* __restrict__ k_in,
    const float* __restrict__ v_in,
    const float* __restrict__ Wq, const float* __restrict__ bq,
    const float* __restrict__ Wk, const float* __restrict__ bk,
    const float* __restrict__ Wv, const float* __restrict__ bv,
    __bf16* __restrict__ Qb, __bf16* __restrict__ Kb, __bf16* __restrict__ Vt) {
  const int wv = threadIdx.x >> 5, ln = threadIdx.x & 31;
  const int hf = ln >> 4, lm = ln & 15;
  const int m0 = blockIdx.x * 64 + wv * 16;
  const int n0 = blockIdx.y * 64;
  const int z  = blockIdx.z;
  const float* X    = (z == 0) ? q_in : (z == 1) ? k_in : v_in;
  const float* W    = (z == 0) ? Wq   : (z == 1) ? Wk   : Wv;
  const float* bias = (z == 0) ? bq   : (z == 1) ? bk   : bv;

  v8f acc[4] = {};
  const float* arow = X + (size_t)(m0 + lm) * Dm;
  for (int k0 = 0; k0 < Dm; k0 += 32) {
    v16bf a = ld_op_f32(arow + k0 + hf * 8);
#pragma unroll
    for (int t = 0; t < 4; ++t) {
      v16bf b = ld_op_f32(W + (size_t)(n0 + t * 16 + lm) * Dm + k0 + hf * 8);
      acc[t] = wmma_bf16(a, b, acc[t]);
    }
  }
#pragma unroll
  for (int t = 0; t < 4; ++t) {
    const int col = n0 + t * 16 + lm;
    const int h = col >> 6, d = col & 63;
    const float bb = bias[col];
#pragma unroll
    for (int e = 0; e < 8; ++e) {
      const int row = m0 + e + 8 * hf;          // global m = b*S + s
      const int bi = row >> 12, s = row & 4095;
      const float v = acc[t][e] + bb;
      if (z == 2) {
        Vt[((size_t)(bi * Hn + h) * HDm + d) * Sq + s] = (__bf16)v;
      } else {
        __bf16* dst = (z == 0) ? Qb : Kb;
        dst[((size_t)(bi * Hn + h) * Sq + s) * HDm + d] = (__bf16)v;
      }
    }
  }
}

// ---------------- Kernel 2: scores = scale * Q K^T (raw, into probs region) --
// grid: (Sq/64, BHn), block 128. K tiles (64x64 bf16, 8KB) staged into LDS by
// the Tensor Data Mover once per workgroup (shared by all 4 waves).
__global__ void __launch_bounds__(128) scores_kernel(
    const __bf16* __restrict__ Qb, const __bf16* __restrict__ Kb,
    float* __restrict__ scores, float* __restrict__ rowmax) {
  __shared__ __align__(16) __bf16 ktile[64 * HDm];
  const int wv = threadIdx.x >> 5, ln = threadIdx.x & 31;
  const int hf = ln >> 4, lm = ln & 15;
  const int bh = blockIdx.y;
  const int q0 = blockIdx.x * 64 + wv * 16;
  const __bf16* Q = Qb + (size_t)bh * Sq * HDm;
  const __bf16* K = Kb + (size_t)bh * Sq * HDm;
  float* sc = scores + (size_t)bh * Sq * Sq;

  const __bf16* qrow = Q + (size_t)(q0 + lm) * HDm;
  const v16bf a0 = ld_op_bf16(qrow + hf * 8);        // d = 0..31
  const v16bf a1 = ld_op_bf16(qrow + 32 + hf * 8);   // d = 32..63

  float vmax[8];
#pragma unroll
  for (int e = 0; e < 8; ++e) vmax[e] = -3.0e38f;

  for (int kc = 0; kc < Sq; kc += 64) {
    __syncthreads();                         // previous tile fully consumed
#if HAVE_TDM
    if (wv == 0) {
      tdm_load_2d(lds_u32(ktile), K + (size_t)kc * HDm,
                  /*tile_d0=*/HDm, /*tile_d1=*/64,
                  /*tensor_d0=*/HDm, /*tensor_d1=*/Sq, /*stride0=*/HDm);
      wait_tensorcnt0();
    }
#else
    {  // cooperative fallback: 64 contiguous K rows = 8KB linear copy
      const v8bf* src = (const v8bf*)(K + (size_t)kc * HDm);
      v8bf* dst = (v8bf*)ktile;
      for (int i = threadIdx.x; i < 512; i += 128) dst[i] = src[i];
    }
#endif
    __syncthreads();                         // tile visible to all waves
#pragma unroll
    for (int kt = 0; kt < 4; ++kt) {
      const __bf16* kr = ktile + (size_t)(kt * 16 + lm) * HDm;
      v16bf b0 = ld_op_bf16(kr + hf * 8);
      v16bf b1 = ld_op_bf16(kr + 32 + hf * 8);
      v8f acc = {};
      acc = wmma_bf16(a0, b0, acc);
      acc = wmma_bf16(a1, b1, acc);
      const int kcc = kc + kt * 16;
#pragma unroll
      for (int e = 0; e < 8; ++e) {
        const float v = acc[e] * 0.125f;            // 1/sqrt(64)
        const int row = q0 + e + 8 * hf;
        sc[(size_t)row * Sq + kcc + lm] = v;        // 16 lanes -> 64B contiguous
        vmax[e] = fmaxf(vmax[e], v);
      }
    }
  }
  // reduce max across the 16 lanes of each half (masks stay within the half)
#pragma unroll
  for (int off = 1; off < 16; off <<= 1) {
#pragma unroll
    for (int e = 0; e < 8; ++e)
      vmax[e] = fmaxf(vmax[e], __shfl_xor(vmax[e], off, 32));
  }
  if (lm == 0) {
#pragma unroll
    for (int e = 0; e < 8; ++e)
      rowmax[(size_t)bh * Sq + q0 + e + 8 * hf] = vmax[e];
  }
}

// ---------------- Kernel 3: row sums of exp(s - m) ----------------
__global__ void __launch_bounds__(256) rowsum_kernel(
    const float* __restrict__ scores, const float* __restrict__ rowmax,
    float* __restrict__ rowsum) {
  const int wv = threadIdx.x >> 5, ln = threadIdx.x & 31;
  const size_t row = (size_t)blockIdx.x * 8 + wv;
  const float m = rowmax[row];
  const float* p = scores + row * Sq;
  float s = 0.f;
  for (int i = ln * 4; i < Sq; i += 128) {
    v4f x = *(const v4f*)(p + i);
    s += __expf(x[0] - m) + __expf(x[1] - m) + __expf(x[2] - m) + __expf(x[3] - m);
  }
#pragma unroll
  for (int off = 1; off < 32; off <<= 1) s += __shfl_xor(s, off, 32);
  if (ln == 0) rowsum[row] = s;
}

// ---------------- Kernel 4: finalize probs + ctx = P @ V ----------------
// grid: (Sq/64, BHn), block 128. V^T tiles (64 d-rows x 64 k-cols bf16, 8KB)
// staged via TDM once per workgroup; shared by all 4 waves.
__global__ void __launch_bounds__(128) attnv_kernel(
    float* __restrict__ scores, const float* __restrict__ rowmax,
    const float* __restrict__ rowsum, const __bf16* __restrict__ Vt,
    __bf16* __restrict__ ctx) {
  __shared__ __align__(16) __bf16 vtile[HDm * 64];
  const int wv = threadIdx.x >> 5, ln = threadIdx.x & 31;
  const int hf = ln >> 4, lm = ln & 15;
  const int bh = blockIdx.y;
  const int b = bh / Hn, h = bh % Hn;
  const int q0 = blockIdx.x * 64 + wv * 16;
  const int row = q0 + lm;                     // A-operand row for this lane
  float* srow = scores + ((size_t)bh * Sq + row) * Sq;
  const float m   = rowmax[(size_t)bh * Sq + row];
  const float inv = 1.0f / rowsum[(size_t)bh * Sq + row];
  const __bf16* Vh = Vt + (size_t)bh * HDm * Sq;

  v8f acc[4] = {};
  for (int kc = 0; kc < Sq; kc += 64) {
    __syncthreads();
#if HAVE_TDM
    if (wv == 0) {
      tdm_load_2d(lds_u32(vtile), Vh + kc,
                  /*tile_d0=*/64, /*tile_d1=*/HDm,
                  /*tensor_d0=*/Sq, /*tensor_d1=*/HDm, /*stride0=*/Sq);
      wait_tensorcnt0();
    }
#else
    {  // cooperative fallback: 64 rows (stride Sq) x 64 cols
      for (int i = threadIdx.x; i < 512; i += 128) {
        const int d = i >> 3, c = (i & 7) * 8;
        *(v8bf*)(vtile + d * 64 + c) = *(const v8bf*)(Vh + (size_t)d * Sq + kc + c);
      }
    }
#endif
    __syncthreads();
#pragma unroll
    for (int sub = 0; sub < 2; ++sub) {
      const int ks = kc + sub * 32;
      float* p = srow + ks + hf * 8;
      __builtin_prefetch(p + 256, 0, 1);        // global_prefetch_b8 (gfx1250)
      v4f x0 = *(const v4f*)p,        x1 = *(const v4f*)(p + 4);
      v4f x2 = *(const v4f*)(p + 16), x3 = *(const v4f*)(p + 20);
      v4f e0, e1, e2, e3;
#pragma unroll
      for (int i = 0; i < 4; ++i) {
        e0[i] = __expf(x0[i] - m) * inv;
        e1[i] = __expf(x1[i] - m) * inv;
        e2[i] = __expf(x2[i] - m) * inv;
        e3[i] = __expf(x3[i] - m) * inv;
      }
      // write final probs back in place (fp32)
      *(v4f*)p = e0; *(v4f*)(p + 4) = e1; *(v4f*)(p + 16) = e2; *(v4f*)(p + 20) = e3;
      // bf16 WMMA A operand
      v16bf a;
#pragma unroll
      for (int i = 0; i < 4; ++i) {
        a[i]      = (__bf16)e0[i];
        a[i + 4]  = (__bf16)e1[i];
        a[i + 8]  = (__bf16)e2[i];
        a[i + 12] = (__bf16)e3[i];
      }
#pragma unroll
      for (int t = 0; t < 4; ++t) {
        v16bf bb = ld_op_bf16(vtile + (size_t)(t * 16 + lm) * 64 + sub * 32 + hf * 8);
        acc[t] = wmma_bf16(a, bb, acc[t]);
      }
    }
  }
#pragma unroll
  for (int t = 0; t < 4; ++t) {
    const int d = t * 16 + lm;
#pragma unroll
    for (int e = 0; e < 8; ++e) {
      const int q = q0 + e + 8 * hf;
      ctx[((size_t)b * Sq + q) * Dm + h * HDm + d] = (__bf16)acc[t][e];
    }
  }
}

// ---------------- Kernel 5: out = ctx @ Wo^T + bo ----------------
__global__ void __launch_bounds__(128) oproj_kernel(
    const __bf16* __restrict__ ctx, const float* __restrict__ Wo,
    const float* __restrict__ bo, float* __restrict__ out) {
  const int wv = threadIdx.x >> 5, ln = threadIdx.x & 31;
  const int hf = ln >> 4, lm = ln & 15;
  const int m0 = blockIdx.x * 64 + wv * 16;
  const int n0 = blockIdx.y * 64;
  v8f acc[4] = {};
  const __bf16* arow = ctx + (size_t)(m0 + lm) * Dm;
  for (int k0 = 0; k0 < Dm; k0 += 32) {
    v16bf a = ld_op_bf16(arow + k0 + hf * 8);
#pragma unroll
    for (int t = 0; t < 4; ++t) {
      v16bf b = ld_op_f32(Wo + (size_t)(n0 + t * 16 + lm) * Dm + k0 + hf * 8);
      acc[t] = wmma_bf16(a, b, acc[t]);
    }
  }
#pragma unroll
  for (int t = 0; t < 4; ++t) {
    const int col = n0 + t * 16 + lm;
    const float bb = bo[col];
#pragma unroll
    for (int e = 0; e < 8; ++e) {
      const int row = m0 + e + 8 * hf;
      out[(size_t)row * Dm + col] = acc[t][e] + bb;
    }
  }
}

extern "C" void kernel_launch(void* const* d_in, const int* in_sizes, int n_in,
                              void* d_out, int out_size, void* d_ws, size_t ws_size,
                              hipStream_t stream) {
  const float* q_in = (const float*)d_in[0];
  const float* k_in = (const float*)d_in[1];
  const float* v_in = (const float*)d_in[2];
  const float* Wq = (const float*)d_in[3];
  const float* bq = (const float*)d_in[4];
  const float* Wk = (const float*)d_in[5];
  const float* bk = (const float*)d_in[6];
  const float* Wv = (const float*)d_in[7];
  const float* bv = (const float*)d_in[8];
  const float* Wo = (const float*)d_in[9];
  const float* bo = (const float*)d_in[10];

  float* out   = (float*)d_out;                     // [B,S,D]
  float* probs = out + (size_t)Bn * Sq * Dm;        // [B,H,S,S]

  const size_t TENS = (size_t)Bn * Sq * Dm;         // 6,291,456 elems
  __bf16* Qb  = (__bf16*)d_ws;
  __bf16* Kb  = Qb + TENS;
  __bf16* Vt  = Kb + TENS;
  __bf16* ctx = Vt + TENS;
  float* rowmax = (float*)(ctx + TENS);
  float* rowsum = rowmax + (size_t)Bn * Hn * Sq;

  dim3 gA(Mr / 64, Dm / 64, 3);
  qkv_proj_kernel<<<gA, 128, 0, stream>>>(q_in, k_in, v_in, Wq, bq, Wk, bk,
                                          Wv, bv, Qb, Kb, Vt);
  dim3 gB(Sq / 64, BHn);
  scores_kernel<<<gB, 128, 0, stream>>>(Qb, Kb, probs, rowmax);
  rowsum_kernel<<<(Bn * Hn * Sq) / 8, 256, 0, stream>>>(probs, rowmax, rowsum);
  attnv_kernel<<<gB, 128, 0, stream>>>(probs, rowmax, rowsum, Vt, ctx);
  dim3 gD(Mr / 64, Dm / 64);
  oproj_kernel<<<gD, 128, 0, stream>>>(ctx, Wo, bo, out);
}
